// StochasticTwoLayerGCN_4724464026102
// MI455X (gfx1250) — matrix operand order repaired and checked by reading, not compiled, not run on hardware
//
#include <hip/hip_runtime.h>

typedef __attribute__((ext_vector_type(16))) __bf16 v16bf;
typedef __attribute__((ext_vector_type(8)))  __bf16 v8bf;
typedef __attribute__((ext_vector_type(8)))  float  v8f;

#define IN_F   512
#define HID_F  512
#define OUT_F  256
#define NDST0  20000   // fixed by reference (not derivable from in_sizes)

// ---------------------------------------------------------------- zero init
__global__ void gcn_zero_f32(float* __restrict__ p, long n) {
    long i = (long)blockIdx.x * blockDim.x + threadIdx.x;
    long stride = (long)gridDim.x * blockDim.x;
    for (; i < n; i += stride) p[i] = 0.0f;
}

// ---------------------------------------------------------------- degrees
__global__ void gcn_count_deg(const int* __restrict__ idx, float* __restrict__ deg, int E) {
    int i = blockIdx.x * blockDim.x + threadIdx.x;
    if (i < E) atomicAdd(&deg[idx[i]], 1.0f);
}

// ------------------------------------------- edge gather -> scatter-add
// one block (128 threads) per edge; float4 per thread; src scale fused.
__global__ void gcn_scatter(const float4* __restrict__ X,
                            const int* __restrict__ src, const int* __restrict__ dst,
                            const float* __restrict__ outdeg,
                            float* __restrict__ agg, int F4) {
    int e = blockIdx.x;
    int s = src[e];
    int d = dst[e];
    float sc = rsqrtf(fmaxf(outdeg[s], 1.0f));
    int c = threadIdx.x;                       // 0..F4-1
    float4 v = X[(long)s * F4 + c];
    float* base = agg + ((long)d * F4 + c) * 4;
    atomicAdd(base + 0, v.x * sc);
    atomicAdd(base + 1, v.y * sc);
    atomicAdd(base + 2, v.z * sc);
    atomicAdd(base + 3, v.w * sc);
}

// ------------------------------------------- fp32 rows -> scaled bf16 rows
// out[m][k] = bf16( A[m][k] * rsqrt(max(indeg[m],1)) ) ; K fixed at 512.
__global__ void gcn_rows_to_bf16(const float4* __restrict__ A,
                                 const float* __restrict__ indeg,
                                 __bf16* __restrict__ out, long n4) {
    long i = (long)blockIdx.x * blockDim.x + threadIdx.x;   // float4 index
    if (i >= n4) return;
    int row = (int)(i >> 7);                    // 128 float4 per 512-wide row
    float sc = rsqrtf(fmaxf(indeg[row], 1.0f));
    float4 v = A[i];
    __bf16* o = out + i * 4;
    o[0] = (__bf16)(v.x * sc);
    o[1] = (__bf16)(v.y * sc);
    o[2] = (__bf16)(v.z * sc);
    o[3] = (__bf16)(v.w * sc);
}

// ------------------------------------------- W[K,N] fp32 -> Wt[N,K] bf16
__global__ void gcn_wT_to_bf16(const float* __restrict__ W, __bf16* __restrict__ Wt,
                               int K, int N) {
    long i = (long)blockIdx.x * blockDim.x + threadIdx.x;
    if (i >= (long)K * N) return;
    int k = (int)(i / N);
    int n = (int)(i - (long)k * N);
    Wt[(long)n * K + k] = (__bf16)W[i];
}

// ------------------------------------------- WMMA GEMM (bf16 in, f32 acc)
// C[M,N] = relu( A[M,K](bf16,row-major,pre-scaled) @ Wt[N,K]^T + bias )
// block: 256 thr = 8 waves; tile M=16, N=128 (one 16x16 WMMA tile per wave).
// No LDS: fragments loaded straight from global in contiguous k order (k = 16*half + s), same for A and B.
__global__ __launch_bounds__(256)
void gcn_gemm_wmma(const __bf16* __restrict__ A, const __bf16* __restrict__ Wt,
                   const float* __restrict__ bias, float* __restrict__ C,
                   int K, int N) {
    const int wave = threadIdx.x >> 5;
    const int lane = threadIdx.x & 31;
    const int half = lane >> 4;          // 0 or 1
    const int l16  = lane & 15;
    const int m0   = blockIdx.x * 16;
    const int n0   = blockIdx.y * 128 + wave * 16;

    // A fragment: lane (m=l16, kh=half) holds contiguous K-run [16*half,+16) of row m
    const __bf16* aRow = A  + (size_t)(m0 + l16) * K + 16 * half;
    // B fragment: lane (n=l16, kh=half) holds contiguous K-run [16*half,+16) of col n
    const __bf16* bRow = Wt + (size_t)(n0 + l16) * K + 16 * half;

    union V16 { v16bf v; v8bf h[2]; };

    v8f acc = {};
    #pragma unroll 4
    for (int k0 = 0; k0 < K; k0 += 32) {
        V16 ua;
        ua.h[0] = *reinterpret_cast<const v8bf*>(aRow + k0);        // global_load_b128
        ua.h[1] = *reinterpret_cast<const v8bf*>(aRow + k0 + 8);    // global_load_b128
        v16bf bv = *reinterpret_cast<const v16bf*>(bRow + k0);      // 2x global_load_b128
        acc = __builtin_amdgcn_wmma_f32_16x16x32_bf16(
            /*neg_a=*/false, ua.v, /*neg_b=*/false, bv,
            /*c_mod=*/(short)0, acc, /*reuse_a=*/false, /*reuse_b=*/false);
    }

    // D layout: VGPR r, lanes 0-15 -> M=r, N=lane; lanes 16-31 -> M=8+r, N=lane-16
    int n = n0 + l16;
    float bv = bias[n];
    int mh = half << 3;
    #pragma unroll
    for (int r = 0; r < 8; ++r) {
        int m = m0 + r + mh;
        C[(long)m * N + n] = fmaxf(acc[r] + bv, 0.0f);
    }
}

// ---------------------------------------------------------------- launcher
extern "C" void kernel_launch(void* const* d_in, const int* in_sizes, int n_in,
                              void* d_out, int out_size, void* d_ws, size_t ws_size,
                              hipStream_t stream) {
    const float* x    = (const float*)d_in[0];
    const int*   src0 = (const int*)  d_in[1];
    const int*   dst0 = (const int*)  d_in[2];
    const int*   src1 = (const int*)  d_in[3];
    const int*   dst1 = (const int*)  d_in[4];
    const float* W1   = (const float*)d_in[5];
    const float* b1   = (const float*)d_in[6];
    const float* W2   = (const float*)d_in[7];
    const float* b2   = (const float*)d_in[8];

    const int E0     = in_sizes[1];
    const int E1     = in_sizes[3];
    const int n_src0 = in_sizes[0] / IN_F;   // 100000
    const int n_dst0 = NDST0;                // 20000
    const int n_dst1 = out_size / OUT_F;     // 4000

    // ---- workspace layout: [zeroed: degs, agg0, agg1][h][bf16: Abf0, Abf1, Wt1, Wt2]
    float* ws      = (float*)d_ws;
    float* outdeg0 = ws;  ws += n_src0;
    float* indeg0  = ws;  ws += n_dst0;
    float* outdeg1 = ws;  ws += n_dst0;
    float* indeg1  = ws;  ws += n_dst1;
    float* agg0    = ws;  ws += (size_t)n_dst0 * IN_F;
    float* agg1    = ws;  ws += (size_t)n_dst1 * HID_F;
    long   zcount  = (long)(ws - (float*)d_ws);       // degs + agg0 + agg1
    float* h       = ws;  ws += (size_t)n_dst0 * HID_F;   // fully overwritten
    __bf16* bws  = (__bf16*)ws;
    __bf16* Abf0 = bws;  bws += (size_t)n_dst0 * IN_F;
    __bf16* Abf1 = bws;  bws += (size_t)n_dst1 * HID_F;
    __bf16* Wt1  = bws;  bws += (size_t)IN_F * HID_F;
    __bf16* Wt2  = bws;  bws += (size_t)HID_F * OUT_F;

    gcn_zero_f32<<<2048, 256, 0, stream>>>((float*)d_ws, zcount);

    // weight transposes are independent of everything else
    gcn_wT_to_bf16<<<(IN_F * HID_F + 255) / 256, 256, 0, stream>>>(W1, Wt1, IN_F, HID_F);
    gcn_wT_to_bf16<<<(HID_F * OUT_F + 255) / 256, 256, 0, stream>>>(W2, Wt2, HID_F, OUT_F);

    gcn_count_deg<<<(E0 + 255) / 256, 256, 0, stream>>>(src0, outdeg0, E0);
    gcn_count_deg<<<(E0 + 255) / 256, 256, 0, stream>>>(dst0, indeg0,  E0);
    gcn_count_deg<<<(E1 + 255) / 256, 256, 0, stream>>>(src1, outdeg1, E1);
    gcn_count_deg<<<(E1 + 255) / 256, 256, 0, stream>>>(dst1, indeg1,  E1);

    // ---- layer 1: aggregate -> scale+cvt -> WMMA GEMM (+bias, relu) -> h
    gcn_scatter<<<E0, IN_F / 4, 0, stream>>>((const float4*)x, src0, dst0, outdeg0,
                                             agg0, IN_F / 4);
    {
        long n4 = (long)n_dst0 * IN_F / 4;
        gcn_rows_to_bf16<<<(int)((n4 + 255) / 256), 256, 0, stream>>>(
            (const float4*)agg0, indeg0, Abf0, n4);
    }
    gcn_gemm_wmma<<<dim3(n_dst0 / 16, HID_F / 128), 256, 0, stream>>>(
        Abf0, Wt1, b1, h, IN_F, HID_F);

    // ---- layer 2: aggregate -> scale+cvt -> WMMA GEMM (+bias, relu) -> out
    gcn_scatter<<<E1, HID_F / 4, 0, stream>>>((const float4*)h, src1, dst1, outdeg1,
                                              agg1, HID_F / 4);
    {
        long n4 = (long)n_dst1 * HID_F / 4;
        gcn_rows_to_bf16<<<(int)((n4 + 255) / 256), 256, 0, stream>>>(
            (const float4*)agg1, indeg1, Abf1, n4);
    }
    gcn_gemm_wmma<<<dim3(n_dst1 / 16, OUT_F / 128), 256, 0, stream>>>(
        Abf1, Wt2, b2, (float*)d_out, HID_F, OUT_F);
}
